// REDecoder_55422257988413
// MI455X (gfx1250) — matrix operand order repaired and checked by reading, not compiled, not run on hardware
//
#include <hip/hip_runtime.h>
#include <hip/hip_bf16.h>
#include <cstddef>
#include <cstdint>

// ---------------- problem constants ----------------
#define BB   16
#define SS   512
#define DD   768
#define EE   128
#define RR   2048
#define HH   768
#define H2   384
#define MM   (BB * RR)          // 32768 rows
#define K1   (2 * DD)           // 1536
#define NOUT 2

typedef __attribute__((ext_vector_type(16))) __bf16 v16bf;
typedef __attribute__((ext_vector_type(8)))  __bf16 v8bf;
typedef __attribute__((ext_vector_type(8)))  float  v8f;

#define LDA 40   // 32 halves + pad -> 80B row stride (16B aligned, conflict-free banks)
#define LDB 40

// ---------------- prep kernels ----------------
__global__ void f32_to_bf16_k(const float* __restrict__ s, __bf16* __restrict__ d, int n) {
    int i = blockIdx.x * 256 + threadIdx.x;
    if (i < n) d[i] = (__bf16)s[i];
}

// s is [K][N] row-major (fan_in, fan_out); d is [N][K] bf16
__global__ void transpose_to_bf16_k(const float* __restrict__ s, __bf16* __restrict__ d, int K, int N) {
    int i = blockIdx.x * 256 + threadIdx.x;
    if (i < K * N) {
        int k = i / N, n = i - k * N;
        d[(size_t)n * K + k] = (__bf16)s[i];
    }
}

// bil [2][384][384] (o,i,j) -> d [(o*384+j)][384] over i  (B matrix [N][K], K=i)
__global__ void bil_to_bf16_k(const float* __restrict__ s, __bf16* __restrict__ d) {
    int i = blockIdx.x * 256 + threadIdx.x;
    if (i < NOUT * H2 * H2) {
        int o = i / (H2 * H2);
        int rem = i - o * H2 * H2;
        int ii = rem / H2, j = rem - ii * H2;
        d[(size_t)(o * H2 + j) * H2 + ii] = (__bf16)s[i];
    }
}

// ---------------- async-to-LDS helpers (CDNA5) ----------------
__device__ __forceinline__ void async_b128(uint32_t lds_addr, const void* gaddr) {
    asm volatile("global_load_async_to_lds_b128 %0, %1, off"
                 :: "v"(lds_addr), "v"((uint64_t)(uintptr_t)gaddr)
                 : "memory");
}
__device__ __forceinline__ void wait_async_le3() {
    asm volatile("s_wait_asynccnt 0x3" ::: "memory");
}
__device__ __forceinline__ void wait_async_0() {
    asm volatile("s_wait_asynccnt 0x0" ::: "memory");
}

// ---------------- WMMA GEMM ----------------
// C[m][n] = act( sum_k A[m][k] * Bw[n][k] + bias[n] )
// GATHER: A row m = concat(hidden[b, ent_start[b, rel[b,r]], :], emb[ent_label[...], :])
template <bool GATHER, bool RELU, bool OUTF32>
__global__ __launch_bounds__(256)
void gemm_wmma_k(const __bf16* __restrict__ Abase,   // GATHER ? hidden_bf16 : A [M][K]
                 const __bf16* __restrict__ embB,
                 const int*    __restrict__ rel,
                 const int*    __restrict__ ent_start,
                 const int*    __restrict__ ent_label,
                 const __bf16* __restrict__ Bw,      // [N][K] bf16
                 const float*  __restrict__ bias,    // [N] or nullptr
                 void*         __restrict__ outp,    // [M][N] bf16 or f32
                 int N, int K)
{
    __shared__ __bf16 As[2][128 * LDA];
    __shared__ __bf16 Bs[2][64 * LDB];
    __shared__ int rowOffH[128];
    __shared__ int rowOffE[128];

    const int tid  = threadIdx.x;
    const int lane = tid & 31;
    const int wave = tid >> 5;
    const int m0   = blockIdx.x * 128;
    const int n0   = blockIdx.y * 64;
    const int wm   = wave & 3;   // 4 waves along M
    const int wn   = wave >> 2;  // 2 waves along N

    if (GATHER) {
        if (tid < 128) {
            int m = m0 + tid;
            int b = m >> 11;          // / RR
            int r = m & (RR - 1);
            int e  = rel[b * RR + r];
            int st = ent_start[b * EE + e];
            int lb = ent_label[b * EE + e];
            rowOffH[tid] = (b * SS + st) * DD;
            rowOffE[tid] = lb * DD;
        }
        __syncthreads();
    }

    // -------- per-thread staging assignment (fixed for whole kernel) --------
    const int ar0 = tid >> 2;          // A tile row 0..63
    const int ar1 = ar0 + 64;          // A tile row 64..127
    const int seg = tid & 3;           // 16B segment within 32-half row
    const int bn  = tid >> 2;          // B tile row (n) 0..63

    // LDS destinations (byte addresses, addrspace(3) offsets) for buffer 0
    const uint32_t ldsA0 = (uint32_t)(uintptr_t)&As[0][ar0 * LDA + seg * 8];
    const uint32_t ldsA1 = (uint32_t)(uintptr_t)&As[0][ar1 * LDA + seg * 8];
    const uint32_t ldsB  = (uint32_t)(uintptr_t)&Bs[0][bn  * LDB + seg * 8];
    const uint32_t bufAo = (uint32_t)((uintptr_t)&As[1][0] - (uintptr_t)&As[0][0]);
    const uint32_t bufBo = (uint32_t)((uintptr_t)&Bs[1][0] - (uintptr_t)&Bs[0][0]);

    // Per-thread global source pointers (advance by 32 halves per stage)
    int eo0 = 0, eo1 = 0;
    const __bf16* aSrc0;
    const __bf16* aSrc1;
    if (GATHER) {
        aSrc0 = Abase + rowOffH[ar0] + seg * 8;
        aSrc1 = Abase + rowOffH[ar1] + seg * 8;
        eo0 = rowOffE[ar0];
        eo1 = rowOffE[ar1];
    } else {
        aSrc0 = Abase + (size_t)(m0 + ar0) * K + seg * 8;
        aSrc1 = Abase + (size_t)(m0 + ar1) * K + seg * 8;
    }
    const __bf16* bSrc = Bw + (size_t)(n0 + bn) * K + seg * 8;

    auto stage = [&](int buf) {
        uint32_t oA = buf ? bufAo : 0u;
        uint32_t oB = buf ? bufBo : 0u;
        async_b128(ldsA0 + oA, aSrc0);
        async_b128(ldsA1 + oA, aSrc1);
        async_b128(ldsB  + oB, bSrc);
    };
    auto advance = [&](int k0next) {   // set pointers for the NEXT stage at k0next
        if (GATHER && k0next == DD) {
            aSrc0 = embB + eo0 + seg * 8;
            aSrc1 = embB + eo1 + seg * 8;
        } else {
            aSrc0 += 32;
            aSrc1 += 32;
        }
        bSrc += 32;
    };

    // fragment LDS offsets (element units), fixed per lane
    const int arow = lane & 15;
    const int akb  = (lane >> 4) * 8;    // A: K[akb..akb+7], K[akb+16..akb+23]
    const int bcol = lane & 15;
    const int bkb  = (lane >> 4) * 8;    // B: K[bkb..bkb+7], K[bkb+16..bkb+23]
    const int aoff0 = (wm * 32 + arow) * LDA + akb;
    const int aoff1 = (wm * 32 + 16 + arow) * LDA + akb;
    const int boff0 = (wn * 32 + bcol) * LDB + bkb;
    const int boff1 = (wn * 32 + 16 + bcol) * LDB + bkb;

    v8f acc[2][2] = {};
    const int nK = K >> 5;

    stage(0);
    advance(32);

    for (int kt = 0; kt < nK; ++kt) {
        const int cur = kt & 1;
        if (kt + 1 < nK) {
            stage((kt + 1) & 1);          // prefetch next tile into other buffer
            advance((kt + 2) * 32);
            wait_async_le3();             // previous stage (3 ops) complete
        } else {
            wait_async_0();
        }
        __syncthreads();                  // all waves' current tile visible in LDS

        v16bf afrag[2], bfrag[2];
        {
            union { v16bf v; v8bf h[2]; } u;
            u.h[0] = *(const v8bf*)&As[cur][aoff0];
            u.h[1] = *(const v8bf*)&As[cur][aoff0 + 16];
            afrag[0] = u.v;
            u.h[0] = *(const v8bf*)&As[cur][aoff1];
            u.h[1] = *(const v8bf*)&As[cur][aoff1 + 16];
            afrag[1] = u.v;
            u.h[0] = *(const v8bf*)&Bs[cur][boff0];
            u.h[1] = *(const v8bf*)&Bs[cur][boff0 + 16];
            bfrag[0] = u.v;
            u.h[0] = *(const v8bf*)&Bs[cur][boff1];
            u.h[1] = *(const v8bf*)&Bs[cur][boff1 + 16];
            bfrag[1] = u.v;
        }
        __syncthreads();                  // reads done before this buffer is re-staged

        #pragma unroll
        for (int i = 0; i < 2; ++i)
            #pragma unroll
            for (int j = 0; j < 2; ++j)
                acc[i][j] = __builtin_amdgcn_wmma_f32_16x16x32_bf16(
                    false, afrag[i], false, bfrag[j], (short)0, acc[i][j], false, false);
    }

    // ---- epilogue: C/D layout -> lane<16: N=lane, M=vgpr r; lane>=16: N=lane-16, M=8+r ----
    #pragma unroll
    for (int i = 0; i < 2; ++i) {
        int mb = m0 + wm * 32 + i * 16 + (lane >> 4) * 8;
        #pragma unroll
        for (int j = 0; j < 2; ++j) {
            int nb = n0 + wn * 32 + j * 16 + (lane & 15);
            float bv = bias ? bias[nb] : 0.0f;
            #pragma unroll
            for (int r = 0; r < 8; ++r) {
                float v = acc[i][j][r] + bv;
                if (RELU) v = fmaxf(v, 0.0f);
                if (OUTF32) ((float*)outp)[(size_t)(mb + r) * N + nb] = v;
                else        ((__bf16*)outp)[(size_t)(mb + r) * N + nb] = (__bf16)v;
            }
        }
    }
}

// ---------------- finalize: bilinear contraction + linear + log-softmax + row loss ----------------
__global__ __launch_bounds__(256)
void finalize_k(const float* __restrict__ U,        // [MM][2*H2]  (n = o*H2 + j)
                const __bf16* __restrict__ heads,   // [MM][H2]
                const __bf16* __restrict__ tails,   // [MM][H2]
                const float* __restrict__ lW,       // [2*H2][2]
                const float* __restrict__ lb,       // [2]
                const int*   __restrict__ rel_label,// [MM]
                float* __restrict__ logits,         // [MM][2]
                float* __restrict__ row_loss)       // [MM]
{
    int wave = threadIdx.x >> 5;
    int lane = threadIdx.x & 31;
    int m = blockIdx.x * 8 + wave;

    const float*  u = U     + (size_t)m * (2 * H2);
    const __bf16* h = heads + (size_t)m * H2;
    const __bf16* t = tails + (size_t)m * H2;

    float s0 = 0.f, s1 = 0.f, l0 = 0.f, l1 = 0.f;
    for (int j = lane; j < H2; j += 32) {
        float tv = (float)t[j];
        float hv = (float)h[j];
        s0 += u[j]      * tv;
        s1 += u[H2 + j] * tv;
        l0 += hv * lW[j * 2 + 0] + tv * lW[(H2 + j) * 2 + 0];
        l1 += hv * lW[j * 2 + 1] + tv * lW[(H2 + j) * 2 + 1];
    }
    #pragma unroll
    for (int off = 16; off > 0; off >>= 1) {
        s0 += __shfl_xor(s0, off, 32);
        s1 += __shfl_xor(s1, off, 32);
        l0 += __shfl_xor(l0, off, 32);
        l1 += __shfl_xor(l1, off, 32);
    }
    if (lane == 0) {
        float z0 = s0 + l0 + lb[0];
        float z1 = s1 + l1 + lb[1];
        logits[(size_t)m * 2 + 0] = z0;
        logits[(size_t)m * 2 + 1] = z1;
        float mx  = fmaxf(z0, z1);
        float lse = mx + logf(expf(z0 - mx) + expf(z1 - mx));
        int lbl = rel_label[m];
        row_loss[m] = -((lbl ? z1 : z0) - lse);
    }
}

__global__ void reduce_loss_k(const float* __restrict__ row_loss, float* __restrict__ out, int n) {
    __shared__ float sm[256];
    float s = 0.f;
    for (int i = threadIdx.x; i < n; i += 256) s += row_loss[i];
    sm[threadIdx.x] = s;
    __syncthreads();
    for (int st = 128; st > 0; st >>= 1) {
        if (threadIdx.x < st) sm[threadIdx.x] += sm[threadIdx.x + st];
        __syncthreads();
    }
    if (threadIdx.x == 0) out[0] = sm[0] / (float)n;
}

// ---------------- host launch ----------------
extern "C" void kernel_launch(void* const* d_in, const int* in_sizes, int n_in,
                              void* d_out, int out_size, void* d_ws, size_t ws_size,
                              hipStream_t stream) {
    const float* hidden = (const float*)d_in[0];
    const float* emb    = (const float*)d_in[1];
    const float* hW1 = (const float*)d_in[2];  const float* hb1 = (const float*)d_in[3];
    const float* hW2 = (const float*)d_in[4];  const float* hb2 = (const float*)d_in[5];
    const float* tW1 = (const float*)d_in[6];  const float* tb1 = (const float*)d_in[7];
    const float* tW2 = (const float*)d_in[8];  const float* tb2 = (const float*)d_in[9];
    const float* bil = (const float*)d_in[10];
    const float* lW  = (const float*)d_in[11]; const float* lb  = (const float*)d_in[12];
    const int* ent_start = (const int*)d_in[13];
    const int* ent_label = (const int*)d_in[14];
    const int* rel_head  = (const int*)d_in[15];
    const int* rel_tail  = (const int*)d_in[16];
    const int* rel_label = (const int*)d_in[17];

    float* out = (float*)d_out;   // out[0] = loss, out+1 = logits [MM*2]

    char* ws = (char*)d_ws;
    size_t off = 0;
    auto alloc = [&](size_t bytes) { size_t r = off; off = (off + bytes + 255) & ~(size_t)255; return r; };

    __bf16* hiddenB = (__bf16*)(ws + alloc((size_t)BB * SS * DD * 2));
    __bf16* embB    = (__bf16*)(ws + alloc((size_t)3 * DD * 2));
    __bf16* hW1T    = (__bf16*)(ws + alloc((size_t)HH * K1 * 2));
    __bf16* tW1T    = (__bf16*)(ws + alloc((size_t)HH * K1 * 2));
    __bf16* hW2T    = (__bf16*)(ws + alloc((size_t)H2 * HH * 2));
    __bf16* tW2T    = (__bf16*)(ws + alloc((size_t)H2 * HH * 2));
    __bf16* bilT    = (__bf16*)(ws + alloc((size_t)NOUT * H2 * H2 * 2));
    __bf16* h1      = (__bf16*)(ws + alloc((size_t)MM * HH * 2));
    __bf16* heads   = (__bf16*)(ws + alloc((size_t)MM * H2 * 2));
    __bf16* tails   = (__bf16*)(ws + alloc((size_t)MM * H2 * 2));
    float*  U       = (float*)(ws + alloc((size_t)MM * 2 * H2 * 4));
    float*  rloss   = (float*)(ws + alloc((size_t)MM * 4));

    // --- prep: bf16 conversions / transposes ---
    {
        int n = BB * SS * DD;
        f32_to_bf16_k<<<(n + 255) / 256, 256, 0, stream>>>(hidden, hiddenB, n);
        n = 3 * DD;
        f32_to_bf16_k<<<(n + 255) / 256, 256, 0, stream>>>(emb, embB, n);
        n = K1 * HH;
        transpose_to_bf16_k<<<(n + 255) / 256, 256, 0, stream>>>(hW1, hW1T, K1, HH);
        transpose_to_bf16_k<<<(n + 255) / 256, 256, 0, stream>>>(tW1, tW1T, K1, HH);
        n = HH * H2;
        transpose_to_bf16_k<<<(n + 255) / 256, 256, 0, stream>>>(hW2, hW2T, HH, H2);
        transpose_to_bf16_k<<<(n + 255) / 256, 256, 0, stream>>>(tW2, tW2T, HH, H2);
        n = NOUT * H2 * H2;
        bil_to_bf16_k<<<(n + 255) / 256, 256, 0, stream>>>(bil, bilT);
    }

    dim3 blk(256);
    // --- head tower ---
    gemm_wmma_k<true, true, false><<<dim3(MM / 128, HH / 64), blk, 0, stream>>>(
        hiddenB, embB, rel_head, ent_start, ent_label, hW1T, hb1, (void*)h1, HH, K1);
    gemm_wmma_k<false, true, false><<<dim3(MM / 128, H2 / 64), blk, 0, stream>>>(
        h1, nullptr, nullptr, nullptr, nullptr, hW2T, hb2, (void*)heads, H2, HH);
    // --- tail tower (reuse h1 buffer) ---
    gemm_wmma_k<true, true, false><<<dim3(MM / 128, HH / 64), blk, 0, stream>>>(
        hiddenB, embB, rel_tail, ent_start, ent_label, tW1T, tb1, (void*)h1, HH, K1);
    gemm_wmma_k<false, true, false><<<dim3(MM / 128, H2 / 64), blk, 0, stream>>>(
        h1, nullptr, nullptr, nullptr, nullptr, tW2T, tb2, (void*)tails, H2, HH);
    // --- bilinear: U[m][(o,j)] = sum_i heads[m,i] * bil[o,i,j] ---
    gemm_wmma_k<false, false, true><<<dim3(MM / 128, (2 * H2) / 64), blk, 0, stream>>>(
        heads, nullptr, nullptr, nullptr, nullptr, bilT, nullptr, (void*)U, 2 * H2, H2);
    // --- finalize + loss ---
    finalize_k<<<MM / 8, 256, 0, stream>>>(U, heads, tails, lW, lb, rel_label, out + 1, rloss);
    reduce_loss_k<<<1, 256, 0, stream>>>(rloss, out, MM);
}
